// GAT_V2_72997264163171
// MI455X (gfx1250) — matrix operand order repaired and checked by reading, not compiled, not run on hardware
//
#include <hip/hip_runtime.h>
#include <stdint.h>

// ---------------------------------------------------------------------------
// GATv2 (3 layers) for MI455X / gfx1250.
//  - Node transforms: bf16 WMMA GEMMs (v_wmma_f32_16x16x32_bf16), wave32.
//    Each wave owns a 16x64 output tile (4 accumulators): one A-fragment load
//    feeds 4 WMMAs, B (weights) are L2-resident. Block = 4 waves = full 256
//    output columns for one 16-row strip.
//  - Edge phase: coalesced gathers (8 threads/edge over heads), segment
//    softmax via order-preserving u32 atomicMax + f32 atomicAdd, scatter-add
//    aggregation with per-edge LDS broadcast of the 8 head alphas.
// ---------------------------------------------------------------------------

static constexpr int HN  = 8;    // heads
static constexpr int DH  = 32;   // per-head dim
static constexpr int FW  = 256;  // H*HID = H*OUT
static constexpr int FIN = 128;  // input feature dim

typedef __attribute__((ext_vector_type(16))) __bf16       v16bf;
typedef __attribute__((ext_vector_type(8)))  float        v8f;
typedef __attribute__((ext_vector_type(4)))  unsigned int u32x4;

union FragB16 { u32x4 u[2]; v16bf v; };

__device__ __forceinline__ unsigned short f2bf(float x) {   // RNE f32->bf16
    unsigned int u = __float_as_uint(x);
    u += 0x7fffu + ((u >> 16) & 1u);
    return (unsigned short)(u >> 16);
}
// order-preserving float<->u32 map for atomicMax on floats
__device__ __forceinline__ unsigned int fenc(float x) {
    unsigned int u = __float_as_uint(x);
    return (u & 0x80000000u) ? ~u : (u | 0x80000000u);
}
__device__ __forceinline__ float fdec(unsigned int u) {
    return __uint_as_float((u & 0x80000000u) ? (u & 0x7fffffffu) : ~u);
}

// ---- small utility kernels -------------------------------------------------
__global__ void fill_u32_k(unsigned int* p, unsigned int v, int n) {
    int i = blockIdx.x * blockDim.x + threadIdx.x;
    if (i < n) p[i] = v;
}
__global__ void fill_f32_k(float* p, float v, int n) {
    int i = blockIdx.x * blockDim.x + threadIdx.x;
    if (i < n) p[i] = v;
}
__global__ void cvt_bf16_k(const float* __restrict__ x, unsigned short* __restrict__ y, int n) {
    int i = blockIdx.x * blockDim.x + threadIdx.x;
    if (i < n) y[i] = f2bf(x[i]);
}
// W [K,Nn] f32 (row major)  ->  Wt [Nn,K] bf16 (row major), WMMA-friendly
__global__ void pack_wT_k(const float* __restrict__ W, unsigned short* __restrict__ Wt,
                          int K, int Nn) {
    int t = blockIdx.x * blockDim.x + threadIdx.x;
    if (t >= K * Nn) return;
    int n = t / K, k = t - n * K;
    Wt[t] = f2bf(W[(size_t)k * Nn + n]);
}

// ---- WMMA GEMM: C[M,256] = A[M,K](bf16) * Bt[256,K]^T(bf16), f32 accum -----
// block = 128 threads = 4 waves; wave w computes rows [bx*16,+16) x cols
// [w*64, +64) as four 16x16 WMMA tiles sharing one A fragment per K-step.
// Requires: M % 16 == 0, K % 32 == 0, Nn == 256.
__global__ __launch_bounds__(128)
void gemm_wmma_k(const unsigned short* __restrict__ A,
                 const unsigned short* __restrict__ Bt,
                 float* __restrict__ C, int M, int K, int Nn) {
    const int wave = threadIdx.x >> 5;
    const int lane = threadIdx.x & 31;
    const int g = lane >> 4;      // lane half (0/1)
    const int r = lane & 15;      // row/col within tile
    const int rowBase = blockIdx.x * 16;
    const int colBase = wave * 64;

    const unsigned short* arow = A  + (size_t)(rowBase + r) * K;
    const unsigned short* brow = Bt + (size_t)(colBase + r) * K;

    v8f acc[4] = {{}, {}, {}, {}};
    for (int kk = 0; kk < K; kk += 32) {
        // A 16x32 bf16 fragment: lane half g holds K chunk [16g,+16)
        FragB16 fa;
        fa.u[0] = *(const u32x4*)(arow + kk + 16 * g);
        fa.u[1] = *(const u32x4*)(arow + kk + 16 * g + 8);
        // 4 B 32x16 fragments (cols colBase+16c): lane half g holds K chunk [16g,+16)
        const unsigned short* bk = brow + kk + 16 * g;
        #pragma unroll
        for (int c = 0; c < 4; ++c) {
            FragB16 fb;
            fb.u[0] = *(const u32x4*)(bk + c * 16 * K);
            fb.u[1] = *(const u32x4*)(bk + c * 16 * K + 8);
            acc[c] = __builtin_amdgcn_wmma_f32_16x16x32_bf16(
                /*neg_a=*/false, fa.v, /*neg_b=*/false, fb.v,
                /*c_mod=*/(short)0, acc[c], /*reuse_a=*/false, /*reuse_b=*/false);
        }
    }
    // C/D layout: VGPR i -> (M = i + 8g, N = r)
    #pragma unroll
    for (int c = 0; c < 4; ++c) {
        float* crow = C + (size_t)(rowBase + 8 * g) * Nn + colBase + 16 * c + r;
        #pragma unroll
        for (int i = 0; i < 8; ++i)
            crow[(size_t)i * Nn] = acc[c][i];
    }
}

// ---- edge kernels ----------------------------------------------------------
// score[e,h] = attn[h] . leaky_relu(fs[src[e],h,:] + fd[dst[e],h,:]); also
// running segment max into m (encoded u32).
__global__ void score_k(const float* __restrict__ fs, const float* __restrict__ fd,
                        const float* __restrict__ attn,
                        const int* __restrict__ src, const int* __restrict__ dst,
                        float* __restrict__ S, unsigned int* __restrict__ m, int E) {
    int t = blockIdx.x * blockDim.x + threadIdx.x;
    if (t >= E * HN) return;
    int e = t >> 3, h = t & 7;
    int s = src[e], d = dst[e];
    const float4* ps = (const float4*)(fs + (size_t)s * FW + h * DH);
    const float4* pd = (const float4*)(fd + (size_t)d * FW + h * DH);
    const float4* pa = (const float4*)(attn + h * DH);
    float acc = 0.f;
    #pragma unroll
    for (int i = 0; i < DH / 4; ++i) {
        float4 a = ps[i], b = pd[i], w = pa[i];
        float x0 = a.x + b.x, x1 = a.y + b.y, x2 = a.z + b.z, x3 = a.w + b.w;
        x0 = (x0 > 0.f) ? x0 : 0.2f * x0;
        x1 = (x1 > 0.f) ? x1 : 0.2f * x1;
        x2 = (x2 > 0.f) ? x2 : 0.2f * x2;
        x3 = (x3 > 0.f) ? x3 : 0.2f * x3;
        acc += x0 * w.x + x1 * w.y + x2 * w.z + x3 * w.w;
    }
    S[t] = acc;
    atomicMax(m + (size_t)d * HN + h, fenc(acc));
}

// S[e,h] = exp(S - max[dst]); denom[dst,h] += S
__global__ void expnorm_k(float* __restrict__ S, const unsigned int* __restrict__ m,
                          float* __restrict__ denom, const int* __restrict__ dst, int E) {
    int t = blockIdx.x * blockDim.x + threadIdx.x;
    if (t >= E * HN) return;
    int e = t >> 3, h = t & 7;
    int d = dst[e];
    float mm = fdec(m[(size_t)d * HN + h]);
    float ex = __expf(S[t] - mm);
    S[t] = ex;
    atomicAdd(denom + (size_t)d * HN + h, ex);
}

// rst[dst, j] += alpha[e,h] * fs[src, j]  (j = h*32+k); one edge per block,
// the 8 head alphas are computed once and broadcast through LDS.
__global__ void agg_k(const float* __restrict__ S, const float* __restrict__ denom,
                      const float* __restrict__ fs,
                      const int* __restrict__ src, const int* __restrict__ dst,
                      float* __restrict__ rst) {
    __shared__ float salpha[HN];
    int e = blockIdx.x;
    int j = threadIdx.x;          // 0..255
    int s = src[e], d = dst[e];
    if (j < HN)
        salpha[j] = S[(size_t)e * HN + j] / denom[(size_t)d * HN + j];
    __syncthreads();
    float alpha = salpha[j >> 5];
    atomicAdd(rst + (size_t)d * FW + j, alpha * fs[(size_t)s * FW + j]);
}

// h_next = [relu](rst + res + bias); write f32 (R) and bf16 (for next GEMM)
__global__ void epilogue_k(const float* __restrict__ rst, const float* __restrict__ res,
                           const float* __restrict__ bias,
                           float* __restrict__ hout, unsigned short* __restrict__ hb,
                           int relu, int n) {
    int i = blockIdx.x * blockDim.x + threadIdx.x;
    if (i >= n) return;
    float v = rst[i] + res[i] + bias[i & (FW - 1)];
    if (relu) v = fmaxf(v, 0.f);
    hout[i] = v;
    hb[i] = f2bf(v);
}

// out[n,k] = mean over heads of (rst + res + bias)
__global__ void final_k(const float* __restrict__ rst, const float* __restrict__ res,
                        const float* __restrict__ bias, float* __restrict__ out, int n) {
    int t = blockIdx.x * blockDim.x + threadIdx.x;
    if (t >= n) return;
    int node = t >> 5, k = t & 31;
    float acc = 0.f;
    #pragma unroll
    for (int h = 0; h < HN; ++h) {
        int j = h * DH + k;
        acc += rst[(size_t)node * FW + j] + res[(size_t)node * FW + j] + bias[j];
    }
    out[t] = acc * 0.125f;
}

// ---------------------------------------------------------------------------
extern "C" void kernel_launch(void* const* d_in, const int* in_sizes, int n_in,
                              void* d_out, int out_size, void* d_ws, size_t ws_size,
                              hipStream_t stream) {
    const float* features = (const float*)d_in[0];
    const int*   src      = (const int*)d_in[1];
    const int*   dst      = (const int*)d_in[2];
    const float* Wsrc1 = (const float*)d_in[3];
    const float* Wdst1 = (const float*)d_in[4];
    const float* attn1 = (const float*)d_in[5];
    const float* b1    = (const float*)d_in[6];
    const float* Wres1 = (const float*)d_in[7];
    const float* Wsrc2 = (const float*)d_in[8];
    const float* Wdst2 = (const float*)d_in[9];
    const float* attn2 = (const float*)d_in[10];
    const float* b2    = (const float*)d_in[11];
    const float* Wsrc3 = (const float*)d_in[12];
    const float* Wdst3 = (const float*)d_in[13];
    const float* attn3 = (const float*)d_in[14];
    const float* b3    = (const float*)d_in[15];

    const int N = in_sizes[0] / FIN;   // 50000 (multiple of 16)
    const int E = in_sizes[1];         // 800000

    // ---- workspace carve-out (~260 MB), 256B aligned slices ----
    size_t off = 0;
    auto carve = [&](size_t bytes) -> void* {
        void* p = (char*)d_ws + off;
        off += (bytes + 255) & ~(size_t)255;
        return p;
    };
    unsigned short* hb  = (unsigned short*)carve((size_t)N * FW * 2); // bf16 activations
    float* fs    = (float*)carve((size_t)N * FW * 4);
    float* fd    = (float*)carve((size_t)N * FW * 4);
    float* R     = (float*)carve((size_t)N * FW * 4);  // residual / current h (f32)
    float* rst   = (float*)carve((size_t)N * FW * 4);
    float* S     = (float*)carve((size_t)E * HN * 4);  // scores then exp weights
    unsigned int* m = (unsigned int*)carve((size_t)N * HN * 4);
    float* denom = (float*)carve((size_t)N * HN * 4);
    unsigned short* WtA = (unsigned short*)carve((size_t)FW * FW * 2);
    unsigned short* WtB = (unsigned short*)carve((size_t)FW * FW * 2);
    unsigned short* WtC = (unsigned short*)carve((size_t)FW * FW * 2);

    const int T = 256;
    const int NB_EH  = (E * HN + T - 1) / T;
    const int NB_NF  = (N * FW + T - 1) / T;
    const int NB_NH  = (N * HN + T - 1) / T;
    const unsigned int ENC_NEG_INF = 0x007FFFFFu;  // fenc(-inf)

    auto run_layer = [&](int K, const unsigned short* Abf,
                         const float* Ws, const float* Wd, const float* attn) {
        // pack transposed bf16 weights
        pack_wT_k<<<(K * FW + T - 1) / T, T, 0, stream>>>(Ws, WtA, K, FW);
        pack_wT_k<<<(K * FW + T - 1) / T, T, 0, stream>>>(Wd, WtB, K, FW);
        // node transforms (one 16-row strip per block, 4 waves cover 256 cols)
        gemm_wmma_k<<<N / 16, 128, 0, stream>>>(Abf, WtA, fs, N, K, FW);
        gemm_wmma_k<<<N / 16, 128, 0, stream>>>(Abf, WtB, fd, N, K, FW);
        // softmax state + accumulator init
        fill_u32_k<<<NB_NH, T, 0, stream>>>(m, ENC_NEG_INF, N * HN);
        fill_f32_k<<<NB_NH, T, 0, stream>>>(denom, 0.f, N * HN);
        fill_f32_k<<<NB_NF, T, 0, stream>>>(rst, 0.f, N * FW);
        // edge phase
        score_k<<<NB_EH, T, 0, stream>>>(fs, fd, attn, src, dst, S, m, E);
        expnorm_k<<<NB_EH, T, 0, stream>>>(S, m, denom, dst, E);
        agg_k<<<E, FW, 0, stream>>>(S, denom, fs, src, dst, rst);
    };

    // ================= Layer 1 (IN=128 -> 256, linear residual) =============
    cvt_bf16_k<<<(N * FIN + T - 1) / T, T, 0, stream>>>(features, hb, N * FIN);
    pack_wT_k<<<(FIN * FW + T - 1) / T, T, 0, stream>>>(Wres1, WtC, FIN, FW);
    gemm_wmma_k<<<N / 16, 128, 0, stream>>>(hb, WtC, R, N, FIN, FW); // residual
    run_layer(FIN, hb, Wsrc1, Wdst1, attn1);
    epilogue_k<<<NB_NF, T, 0, stream>>>(rst, R, b1, R, hb, /*relu=*/1, N * FW);

    // ================= Layer 2 (256 -> 256, identity residual) ==============
    run_layer(FW, hb, Wsrc2, Wdst2, attn2);
    epilogue_k<<<NB_NF, T, 0, stream>>>(rst, R, b2, R, hb, /*relu=*/1, N * FW);

    // ================= Layer 3 (256 -> 256, identity residual, head mean) ===
    run_layer(FW, hb, Wsrc3, Wdst3, attn3);
    final_k<<<(N * DH + T - 1) / T, T, 0, stream>>>(rst, R, b3, (float*)d_out, N * DH);
}
